// GAT_45681272160995
// MI455X (gfx1250) — compile-verified, matrix operand order checked by reading, not run on hardware
//
#include <hip/hip_runtime.h>

typedef _Float16 f16;
typedef __attribute__((ext_vector_type(16))) _Float16 v16h;
typedef __attribute__((ext_vector_type(8)))  float    v8f;
typedef __attribute__((ext_vector_type(4)))  int      v4i;

#define IN_CH 128
#define HF    64   // HEADS * OUT_CH
#define HEADS 8
#define FPH   8    // features per head

// ---------------- gfx1250 async global->LDS path (guarded) --------------------
#define AS1 __attribute__((address_space(1)))
#define AS3 __attribute__((address_space(3)))

#if defined(__has_builtin)
#if __has_builtin(__builtin_amdgcn_global_load_async_to_lds_b128)
#define HAVE_ASYNC_LDS 1
#endif
#endif

#if defined(HAVE_ASYNC_LDS)
#if __has_builtin(__builtin_amdgcn_s_wait_asynccnt)
#define WAIT_ASYNC() __builtin_amdgcn_s_wait_asynccnt(0)
#else
#define WAIT_ASYNC() asm volatile("s_wait_asynccnt 0" ::: "memory")
#endif
#endif

// ---------------- init: zero out accumulator, -inf seg-max, 0 seg-sum ---------
__global__ void gat_init_k(float* __restrict__ out, float* __restrict__ m,
                           float* __restrict__ s, int n_out, int n_nh) {
  int i = blockIdx.x * blockDim.x + threadIdx.x;
  if (i < n_out) out[i] = 0.0f;
  if (i < n_nh) { m[i] = -__builtin_inff(); s[i] = 0.0f; }
}

// ---------------- h = x @ W  (f32 -> f16 inputs, f32 accum, WMMA) -------------
// block = 128 threads (4 waves). Block bm covers rows [16*bm, 16*bm+16).
// Wave w computes column tile w (cols 16w..16w+15). K = 128 in 4 chunks of 32.
// W is staged into LDS pre-swizzled into per-lane fragment order so each B
// fragment is one contiguous 32-byte v16h load (2x ds_load_b128).
__global__ void __launch_bounds__(128) gat_gemm_wmma(const float* __restrict__ x,
                                                     const float* __restrict__ W,
                                                     float* __restrict__ h) {
  __shared__ __align__(16) float xs[16 * IN_CH];   // 8 KB raw fp32 x-strip
  __shared__ __align__(16) f16   Ws[IN_CH * HF];   // 16 KB f16 W, fragment order
  const int bm  = blockIdx.x;
  const int tid = threadIdx.x;

  const float* xrow = x + (size_t)bm * 16 * IN_CH;

#if defined(HAVE_ASYNC_LDS)
  // async DMA of the 16x128 fp32 strip: 4 x b128 per thread, ASYNCcnt-tracked
  for (int i = tid * 4; i < 16 * IN_CH; i += 128 * 4)
    __builtin_amdgcn_global_load_async_to_lds_b128((AS1 v4i*)(xrow + i),
                                                   (AS3 v4i*)(xs + i), 0, 0);
#else
  for (int idx = tid; idx < 16 * IN_CH; idx += 128) xs[idx] = xrow[idx];
#endif

  // stage W once per block (coalesced reads, L2-resident), swizzled to
  // fragment order: frag(ko, wtile, lane)[2j+t] contiguous.
  for (int idx = tid; idx < IN_CH * HF; idx += 128) {
    const int K      = idx >> 6;        // 0..127
    const int col    = idx & 63;        // 0..63
    const int wtile  = col >> 4;
    const int n      = col & 15;
    const int ko     = K >> 5;
    const int kin    = K & 31;
    const int half   = (kin >> 3) & 1;
    const int j      = ((kin >> 4) & 1) * 4 + ((kin & 7) >> 1);
    const int t      = kin & 1;
    const int lane   = half * 16 + n;
    Ws[(((ko * 4 + wtile) * 32 + lane) << 4) + 2 * j + t] = (f16)W[idx];
  }

#if defined(HAVE_ASYNC_LDS)
  WAIT_ASYNC();
#endif
  __syncthreads();

  const int wave = tid >> 5;        // column tile 0..3
  const int lane = tid & 31;
  const int half = lane >> 4;       // lane group 0/1
  const int mn   = lane & 15;       // A-frag row / B,C,D-frag col

  v8f c = {};
#pragma unroll
  for (int ko = 0; ko < IN_CH / 32; ++ko) {
    // A fragment: two contiguous 8-float runs from xs, converted to f16
    v16h a;
#pragma unroll
    for (int j = 0; j < 8; ++j) {
      const int kbase = ((j & 4) ? 16 : 0) + 8 * half + (j & 3) * 2;
#pragma unroll
      for (int t = 0; t < 2; ++t)
        a[2 * j + t] = (f16)xs[mn * IN_CH + 32 * ko + kbase + t];
    }
    // B fragment: one contiguous 32-byte LDS read in fragment order
    const v16h b = *(const v16h*)&Ws[((ko * 4 + wave) * 32 + lane) << 4];

    // D = A*B + C  ->  v_wmma_f32_16x16x32_f16
    c = __builtin_amdgcn_wmma_f32_16x16x32_f16(false, a, false, b,
                                               (short)0, c, false, false);
  }

  // D layout: VGPR r -> row r + 8*half, col mn
  float* hout = h + ((size_t)bm * 16 + 8 * half) * HF + wave * 16 + mn;
#pragma unroll
  for (int r = 0; r < 8; ++r) hout[(size_t)r * HF] = c[r];
}

// ---------------- per-node attention logits a_src, a_dst ----------------------
__global__ void gat_node_att_k(const float* __restrict__ h,
                               const float* __restrict__ att_src,
                               const float* __restrict__ att_dst,
                               float* __restrict__ asrc, float* __restrict__ adst,
                               int n_nh) {
  int i = blockIdx.x * blockDim.x + threadIdx.x;
  if (i >= n_nh) return;
  const int node = i >> 3, hd = i & 7;
  const float* hp = h + (size_t)node * HF + hd * FPH;
  const float* ps = att_src + hd * FPH;
  const float* pd = att_dst + hd * FPH;
  float vs = 0.0f, vd = 0.0f;
#pragma unroll
  for (int f = 0; f < FPH; ++f) { float v = hp[f]; vs += v * ps[f]; vd += v * pd[f]; }
  asrc[i] = vs;
  adst[i] = vd;
}

// ---------------- helpers -----------------------------------------------------
__device__ __forceinline__ void edge_nodes(const int* __restrict__ ei, int E,
                                           int idx, int& s, int& d) {
  if (idx < E) { s = ei[idx]; d = ei[E + idx]; }
  else         { s = idx - E; d = idx - E; }      // self loops appended
}

__device__ __forceinline__ float leaky02(float v) { return v > 0.0f ? v : 0.2f * v; }

__device__ __forceinline__ void atomic_max_f32(float* addr, float val) {
  unsigned int* ai = (unsigned int*)addr;
  unsigned int old = *ai;
  while (__uint_as_float(old) < val) {
    unsigned int assumed = old;
    old = atomicCAS(ai, assumed, __float_as_uint(val));
    if (old == assumed) break;
  }
}

// ---------------- pass 1: segment max over destination ------------------------
__global__ void gat_edge_max_k(const int* __restrict__ ei,
                               const float* __restrict__ asrc,
                               const float* __restrict__ adst,
                               float* __restrict__ m, int E, int ET) {
  int i = blockIdx.x * blockDim.x + threadIdx.x;
  if (i >= ET * HEADS) return;
  const int idx = i >> 3, hd = i & 7;
  int s, d; edge_nodes(ei, E, idx, s, d);
  const float e = leaky02(asrc[s * HEADS + hd] + adst[d * HEADS + hd]);
  atomic_max_f32(&m[d * HEADS + hd], e);
}

// ---------------- pass 2: segment sum of exp(e - m) ---------------------------
__global__ void gat_edge_sum_k(const int* __restrict__ ei,
                               const float* __restrict__ asrc,
                               const float* __restrict__ adst,
                               const float* __restrict__ m,
                               float* __restrict__ ssum, int E, int ET) {
  int i = blockIdx.x * blockDim.x + threadIdx.x;
  if (i >= ET * HEADS) return;
  const int idx = i >> 3, hd = i & 7;
  int s, d; edge_nodes(ei, E, idx, s, d);
  const float e  = leaky02(asrc[s * HEADS + hd] + adst[d * HEADS + hd]);
  const float ex = __expf(e - m[d * HEADS + hd]);
  atomicAdd(&ssum[d * HEADS + hd], ex);
}

// ---------------- pass 3: weighted scatter out[dst] += alpha * h[src] ---------
__global__ void gat_edge_scatter_k(const int* __restrict__ ei,
                                   const float* __restrict__ asrc,
                                   const float* __restrict__ adst,
                                   const float* __restrict__ m,
                                   const float* __restrict__ ssum,
                                   const float* __restrict__ h,
                                   float* __restrict__ out, int E, int ET) {
  long long i = (long long)blockIdx.x * blockDim.x + threadIdx.x;
  if (i >= (long long)ET * HF) return;
  const int idx = (int)(i >> 6);
  const int c   = (int)(i & 63);
  const int hd  = c >> 3;
  int s, d; edge_nodes(ei, E, idx, s, d);
  const float e     = leaky02(asrc[s * HEADS + hd] + adst[d * HEADS + hd]);
  const float ex    = __expf(e - m[d * HEADS + hd]);
  const float alpha = ex / (ssum[d * HEADS + hd] + 1e-16f);
  atomicAdd(&out[(size_t)d * HF + c], alpha * h[(size_t)s * HF + c]);
}

// ---------------- pass 4: bias + ELU ------------------------------------------
__global__ void gat_finalize_k(float* __restrict__ out,
                               const float* __restrict__ bias, int n) {
  int i = blockIdx.x * blockDim.x + threadIdx.x;
  if (i >= n) return;
  float v = out[i] + bias[i & (HF - 1)];
  out[i] = v > 0.0f ? v : (__expf(v) - 1.0f);
}

// ==============================================================================
extern "C" void kernel_launch(void* const* d_in, const int* in_sizes, int n_in,
                              void* d_out, int out_size, void* d_ws, size_t ws_size,
                              hipStream_t stream) {
  const float* x       = (const float*)d_in[0];  // [N, 128]
  const float* W       = (const float*)d_in[1];  // [128, 64]
  const float* att_src = (const float*)d_in[2];  // [1, 8, 8]
  const float* att_dst = (const float*)d_in[3];  // [1, 8, 8]
  const float* bias    = (const float*)d_in[4];  // [64]
  const int*   ei      = (const int*)d_in[5];    // [2, E] (int32: jax x64 off)

  const int N  = in_sizes[0] / IN_CH;
  const int E  = in_sizes[5] / 2;
  const int ET = E + N;                          // edges + self loops

  float* out = (float*)d_out;

  // workspace layout: h | a_src | a_dst | seg_max | seg_sum  (N*96 floats)
  float* h    = (float*)d_ws;
  float* asrc = h    + (size_t)N * HF;
  float* adst = asrc + (size_t)N * HEADS;
  float* segm = adst + (size_t)N * HEADS;
  float* segs = segm + (size_t)N * HEADS;

  const int n_out = N * HF;
  const int n_nh  = N * HEADS;

  gat_init_k<<<(n_out + 255) / 256, 256, 0, stream>>>(out, segm, segs, n_out, n_nh);

  gat_gemm_wmma<<<N / 16, 128, 0, stream>>>(x, W, h);

  gat_node_att_k<<<(n_nh + 255) / 256, 256, 0, stream>>>(h, att_src, att_dst,
                                                         asrc, adst, n_nh);

  const int n_eh = ET * HEADS;
  gat_edge_max_k<<<(n_eh + 255) / 256, 256, 0, stream>>>(ei, asrc, adst, segm, E, ET);
  gat_edge_sum_k<<<(n_eh + 255) / 256, 256, 0, stream>>>(ei, asrc, adst, segm, segs, E, ET);

  const long long n_ec = (long long)ET * HF;
  gat_edge_scatter_k<<<(unsigned)((n_ec + 255) / 256), 256, 0, stream>>>(
      ei, asrc, adst, segm, segs, h, out, E, ET);

  gat_finalize_k<<<(n_out + 255) / 256, 256, 0, stream>>>(out, bias, n_out);
}
